// FastestFlashFA4_738734374973
// MI455X (gfx1250) — matrix operand order repaired and checked by reading, not compiled, not run on hardware
//
#include <hip/hip_runtime.h>
#include <hip/hip_bf16.h>
#include <stdint.h>

// ---------------------------------------------------------------------------
// MI455X / gfx1250 sliding-window attention layer (compile-only target).
//   86 GFLOP vs ~100MB HBM traffic -> compute bound -> bf16 WMMA everywhere.
//   GEMMs: 128x128 block tile, double-buffered LDS fed by
//   GLOBAL_LOAD_ASYNC_TO_LDS_B128 (ASYNCcnt), 8 WMMA per wave per k-step.
//   Flash attention: softcap + 1024-window online softmax, WMMA QK^T and PV.
// ---------------------------------------------------------------------------

#define SEQ    2048
#define HID    2048
#define NHEAD  16
#define HDIM   128
#define WIN    1024
#define CAP    30.0f
#define QSCALE 0.08838834764831845f   // 1/sqrt(128)

typedef __attribute__((ext_vector_type(16))) __bf16 v16bf;
typedef __attribute__((ext_vector_type(8)))  __bf16 v8bf;
typedef __attribute__((ext_vector_type(8)))  float  v8f;

union FragA { v16bf v; v8bf h[2]; };

__device__ __forceinline__ uint16_t f32_to_bf16(float f) {
  uint32_t u = __float_as_uint(f);
  uint32_t r = (u + 0x7FFFu + ((u >> 16) & 1u)) >> 16;   // round-nearest-even
  return (uint16_t)r;
}

// ------------------------------ convert ------------------------------------
__global__ void f32_to_bf16_kernel(const float4* __restrict__ in,
                                   uint64_t* __restrict__ out, int n4) {
  int i = blockIdx.x * blockDim.x + threadIdx.x;
  if (i < n4) {
    float4 f = in[i];
    uint64_t p = (uint64_t)f32_to_bf16(f.x) |
                 ((uint64_t)f32_to_bf16(f.y) << 16) |
                 ((uint64_t)f32_to_bf16(f.z) << 32) |
                 ((uint64_t)f32_to_bf16(f.w) << 48);
    out[i] = p;
  }
}

// ------------------------------ GEMM ---------------------------------------
// Y[m,n] = sum_k A[m,k] * B[n,k]   (A row-major, B = weight [out,in])
// mode 0: bf16 out, [H,S,D]  (Q, K; Q pre-scaled)
// mode 1: bf16 out, [H,D,S]  (V transposed so PV B-fragments are contiguous)
// mode 2: f32  out, [S,HID]  (final O projection -> d_out)
#define GST 40   // LDS row stride (uint16), 80B rows => 16B-aligned octets

__device__ __forceinline__ void store_gemm(void* out, int mode, int m, int n,
                                           float v) {
  if (mode == 2) {
    ((float*)out)[m * HID + n] = v;
  } else {
    int h = n >> 7, d = n & 127;
    uint16_t b = f32_to_bf16(v);
    if (mode == 0)
      ((uint16_t*)out)[(h << 18) + (m << 7) + d] = b;   // [H,S,D]
    else
      ((uint16_t*)out)[(h << 18) + (d << 11) + m] = b;  // [H,D,S]
  }
}

// Issue one thread's 32B share of a 128x32 bf16 panel as two async
// LDS-direct loads. INST_OFFSET is added to BOTH lds and global addresses
// (ISA 10. async pseudocode), so one address pair covers both halves.
__device__ __forceinline__ void stage_panel_async(uint32_t lds, uint64_t gaddr) {
  asm volatile("global_load_async_to_lds_b128 %0, %1, off"
               :: "v"(lds), "v"(gaddr) : "memory");
  asm volatile("global_load_async_to_lds_b128 %0, %1, off offset:16"
               :: "v"(lds), "v"(gaddr) : "memory");
}

__global__ __launch_bounds__(256) void gemm_bf16_wmma(
    const uint16_t* __restrict__ A,   // [2048,2048] bf16 bits
    const uint16_t* __restrict__ B,   // [2048,2048] bf16 bits (W[o][i])
    void* __restrict__ out, float out_scale, int mode) {
  __shared__ __align__(16) uint16_t As[2][128 * GST];
  __shared__ __align__(16) uint16_t Bs[2][128 * GST];

  const int tid  = threadIdx.x;
  const int lane = tid & 31;
  const int wave = tid >> 5;
  const int wm   = (wave >> 1) * 32;   // wave rows: 0/32/64/96
  const int wn   = (wave & 1) * 64;    // wave cols: 0/64
  const int bm   = blockIdx.x * 128;
  const int bn   = blockIdx.y * 128;
  const int l15  = lane & 15;
  const int lh   = lane >> 4;

  // staging assignment: thread -> (row, 16-col half) of the 128x32 panel
  const int srow = tid >> 1;
  const int scol = (tid & 1) * 16;
  uint32_t ldsA[2], ldsB[2];
  ldsA[0] = (uint32_t)(uintptr_t)&As[0][srow * GST + scol];
  ldsA[1] = (uint32_t)(uintptr_t)&As[1][srow * GST + scol];
  ldsB[0] = (uint32_t)(uintptr_t)&Bs[0][srow * GST + scol];
  ldsB[1] = (uint32_t)(uintptr_t)&Bs[1][srow * GST + scol];
  const uint64_t gA = (uint64_t)(uintptr_t)(A + (bm + srow) * HID + scol);
  const uint64_t gB = (uint64_t)(uintptr_t)(B + (bn + srow) * HID + scol);

  v8f c[2][4];
#pragma unroll
  for (int mi = 0; mi < 2; ++mi)
#pragma unroll
    for (int ni = 0; ni < 4; ++ni) c[mi][ni] = (v8f){};

  // prime buffer 0
  stage_panel_async(ldsA[0], gA);
  stage_panel_async(ldsB[0], gB);
  asm volatile("s_wait_asynccnt 0" ::: "memory");
  __syncthreads();

  int cur = 0;
  for (int k0 = 0; k0 < HID; k0 += 32) {
    if (k0 + 32 < HID) {   // prefetch next panel into the other buffer
      stage_panel_async(ldsA[cur ^ 1], gA + (uint64_t)(k0 + 32) * 2);
      stage_panel_async(ldsB[cur ^ 1], gB + (uint64_t)(k0 + 32) * 2);
    }

    const uint16_t* as = As[cur];
    const uint16_t* bs = Bs[cur];
    FragA a[2];
#pragma unroll
    for (int mi = 0; mi < 2; ++mi) {
      int am = wm + mi * 16 + l15;
      int kb = lh * 16;
      a[mi].h[0] = *(const v8bf*)&as[am * GST + kb];
      a[mi].h[1] = *(const v8bf*)&as[am * GST + kb + 8];
    }
    FragA b[4];
#pragma unroll
    for (int ni = 0; ni < 4; ++ni) {
      int n0  = wn + ni * 16 + l15;
      int kb2 = lh * 16;
      b[ni].h[0] = *(const v8bf*)&bs[n0 * GST + kb2];
      b[ni].h[1] = *(const v8bf*)&bs[n0 * GST + kb2 + 8];
    }
#pragma unroll
    for (int mi = 0; mi < 2; ++mi)
#pragma unroll
      for (int ni = 0; ni < 4; ++ni)
        c[mi][ni] = __builtin_amdgcn_wmma_f32_16x16x32_bf16(
            false, a[mi].v, false, b[ni].v, (short)0, c[mi][ni], false, false);

    asm volatile("s_wait_asynccnt 0" ::: "memory");
    __syncthreads();
    cur ^= 1;
  }

  // C layout: row = (lane>>4)*8 + r, col = lane&15
#pragma unroll
  for (int mi = 0; mi < 2; ++mi) {
    int row = bm + wm + mi * 16 + lh * 8;
#pragma unroll
    for (int ni = 0; ni < 4; ++ni) {
      int col = bn + wn + ni * 16 + l15;
#pragma unroll
      for (int r = 0; r < 8; ++r)
        store_gemm(out, mode, row + r, col, c[mi][ni][r] * out_scale);
    }
  }
}

// --------------------------- flash attention -------------------------------
// grid (SEQ/64, NHEAD), block 128 (4 waves); each wave owns 16 query rows.
#define PST 72   // per-wave LDS P row stride (uint16), 144B = 16B-aligned

__global__ __launch_bounds__(128) void flash_attn_kernel(
    const uint16_t* __restrict__ Q,   // [H,S,D], pre-scaled by 1/sqrt(D)
    const uint16_t* __restrict__ K,   // [H,S,D]
    const uint16_t* __restrict__ Vt,  // [H,D,S]
    uint16_t* __restrict__ O) {       // [S, H*D] bf16
  __shared__ __align__(16) uint16_t Plds[4 * 16 * PST];

  const int tid  = threadIdx.x;
  const int lane = tid & 31;
  const int wave = tid >> 5;
  const int head = blockIdx.y;
  const int q0   = blockIdx.x * 64 + wave * 16;
  const int l15  = lane & 15;
  const int lh   = lane >> 4;

  const uint16_t* Qh = Q + head * (SEQ * HDIM);
  const uint16_t* Kh = K + head * (SEQ * HDIM);
  const uint16_t* Vh = Vt + head * (HDIM * SEQ);
  uint16_t* Pw = &Plds[wave * 16 * PST];

  // Q fragments: 16x128 as four 16x32 A-fragments
  FragA qf[4];
  {
    int qrow = q0 + l15;
    int kb   = lh * 16;
#pragma unroll
    for (int kd = 0; kd < 4; ++kd) {
      const uint16_t* p = &Qh[qrow * HDIM + kd * 32 + kb];
      qf[kd].h[0] = *(const v8bf*)p;
      qf[kd].h[1] = *(const v8bf*)(p + 8);
    }
  }

  v8f o[8];
#pragma unroll
  for (int i = 0; i < 8; ++i) o[i] = (v8f){};
  float m_r[8], l_r[8];
#pragma unroll
  for (int r = 0; r < 8; ++r) { m_r[r] = -1e30f; l_r[r] = 0.0f; }

  int kt_lo = q0 - (WIN - 1);
  if (kt_lo < 0) kt_lo = 0;
  kt_lo >>= 6;
  const int kt_hi = (q0 + 15) >> 6;

  for (int kt = kt_lo; kt <= kt_hi; ++kt) {
    const int kbase = kt * 64;

    // ---- S = Q K^T : 16x64 in four 16x16 accumulators ----
    v8f s[4];
#pragma unroll
    for (int j = 0; j < 4; ++j) s[j] = (v8f){};
#pragma unroll
    for (int j = 0; j < 4; ++j) {
      int key = kbase + j * 16 + l15;
#pragma unroll
      for (int kd = 0; kd < 4; ++kd) {
        FragA kf;  // B fragment: lane half picks 16 contiguous features
        const uint16_t* p = &Kh[key * HDIM + kd * 32 + lh * 16];
        kf.h[0] = *(const v8bf*)p;
        kf.h[1] = *(const v8bf*)(p + 8);
        s[j] = __builtin_amdgcn_wmma_f32_16x16x32_bf16(
            false, qf[kd].v, false, kf.v, (short)0, s[j], false, false);
      }
    }

    // ---- softcap + causal/window mask + tile row max ----
    float tm[8];
#pragma unroll
    for (int r = 0; r < 8; ++r) tm[r] = -1e30f;
#pragma unroll
    for (int j = 0; j < 4; ++j) {
      int key = kbase + j * 16 + l15;
#pragma unroll
      for (int r = 0; r < 8; ++r) {
        int q   = q0 + lh * 8 + r;
        float x = s[j][r];
        x = CAP * tanhf(x * (1.0f / CAP));
        bool ok = (key <= q) && (q - key < WIN);
        x = ok ? x : -1e30f;
        s[j][r] = x;
        tm[r] = fmaxf(tm[r], x);
      }
    }
#pragma unroll
    for (int r = 0; r < 8; ++r) {   // reduce over 16 lanes within half
      float t = tm[r];
      t = fmaxf(t, __shfl_xor(t, 1, 32));
      t = fmaxf(t, __shfl_xor(t, 2, 32));
      t = fmaxf(t, __shfl_xor(t, 4, 32));
      t = fmaxf(t, __shfl_xor(t, 8, 32));
      tm[r] = t;
    }

    // ---- online softmax ----
    float corr[8];
#pragma unroll
    for (int r = 0; r < 8; ++r) {
      float mn = fmaxf(m_r[r], tm[r]);
      corr[r]  = __expf(m_r[r] - mn);
      m_r[r]   = mn;
    }
    float rs[8];
#pragma unroll
    for (int r = 0; r < 8; ++r) rs[r] = 0.0f;
#pragma unroll
    for (int j = 0; j < 4; ++j) {
#pragma unroll
      for (int r = 0; r < 8; ++r) {
        float x = s[j][r];
        float p = (x > -1e29f) ? __expf(x - m_r[r]) : 0.0f;
        s[j][r] = p;
        rs[r] += p;
      }
    }
#pragma unroll
    for (int r = 0; r < 8; ++r) {
      float t = rs[r];
      t += __shfl_xor(t, 1, 32);
      t += __shfl_xor(t, 2, 32);
      t += __shfl_xor(t, 4, 32);
      t += __shfl_xor(t, 8, 32);
      l_r[r] = l_r[r] * corr[r] + t;
    }
#pragma unroll
    for (int dt = 0; dt < 8; ++dt)
#pragma unroll
      for (int r = 0; r < 8; ++r) o[dt][r] *= corr[r];

    // ---- relayout P: C layout -> LDS row-major -> A fragments ----
#pragma unroll
    for (int j = 0; j < 4; ++j)
#pragma unroll
      for (int r = 0; r < 8; ++r)
        Pw[(lh * 8 + r) * PST + j * 16 + l15] = f32_to_bf16(s[j][r]);
    asm volatile("s_wait_dscnt 0" ::: "memory");  // same-wave LDS RAW

    FragA pf[2];
    {
      int kb = lh * 16;
#pragma unroll
      for (int t = 0; t < 2; ++t) {
        const uint16_t* p = &Pw[l15 * PST + t * 32 + kb];
        pf[t].h[0] = *(const v8bf*)p;
        pf[t].h[1] = *(const v8bf*)(p + 8);
      }
    }

    // ---- O += P V : eight d-chunks of 16, two 32-key steps ----
#pragma unroll
    for (int dt = 0; dt < 8; ++dt) {
      int d = dt * 16 + l15;
#pragma unroll
      for (int t = 0; t < 2; ++t) {
        FragA vf;  // B fragment from V^T: contiguous over keys
        const uint16_t* p = &Vh[d * SEQ + kbase + t * 32 + lh * 16];
        vf.h[0] = *(const v8bf*)p;
        vf.h[1] = *(const v8bf*)(p + 8);
        o[dt] = __builtin_amdgcn_wmma_f32_16x16x32_bf16(
            false, pf[t].v, false, vf.v, (short)0, o[dt], false, false);
      }
    }
  }

  // ---- normalize and store bf16 [S, H*D] ----
#pragma unroll
  for (int r = 0; r < 8; ++r) {
    int q = q0 + lh * 8 + r;
    float inv = 1.0f / l_r[r];
#pragma unroll
    for (int dt = 0; dt < 8; ++dt) {
      int col = head * HDIM + dt * 16 + l15;
      O[q * HID + col] = f32_to_bf16(o[dt][r] * inv);
    }
  }
}

// ------------------------------ launch -------------------------------------
extern "C" void kernel_launch(void* const* d_in, const int* in_sizes, int n_in,
                              void* d_out, int out_size, void* d_ws,
                              size_t ws_size, hipStream_t stream) {
  const float* x  = (const float*)d_in[0];
  const float* wq = (const float*)d_in[1];
  const float* wk = (const float*)d_in[2];
  const float* wv = (const float*)d_in[3];
  const float* wo = (const float*)d_in[4];

  uint8_t* ws = (uint8_t*)d_ws;
  const size_t MB = 1u << 20;
  uint16_t* xb  = (uint16_t*)(ws + 0 * MB);
  uint16_t* wqb = (uint16_t*)(ws + 8 * MB);
  uint16_t* wkb = (uint16_t*)(ws + 16 * MB);
  uint16_t* wvb = (uint16_t*)(ws + 24 * MB);
  uint16_t* wob = (uint16_t*)(ws + 32 * MB);
  uint16_t* Qb  = (uint16_t*)(ws + 40 * MB);  // [H,S,D] bf16, pre-scaled
  uint16_t* Kb  = (uint16_t*)(ws + 48 * MB);  // [H,S,D] bf16
  uint16_t* Vt  = (uint16_t*)(ws + 56 * MB);  // [H,D,S] bf16
  uint16_t* Ab  = (uint16_t*)(ws + 64 * MB);  // [S,HID] bf16 attention out

  const int n4 = (SEQ * HID) / 4;
  dim3 cblk(256), cgrd((n4 + 255) / 256);
  f32_to_bf16_kernel<<<cgrd, cblk, 0, stream>>>((const float4*)x, (uint64_t*)xb, n4);
  f32_to_bf16_kernel<<<cgrd, cblk, 0, stream>>>((const float4*)wq, (uint64_t*)wqb, n4);
  f32_to_bf16_kernel<<<cgrd, cblk, 0, stream>>>((const float4*)wk, (uint64_t*)wkb, n4);
  f32_to_bf16_kernel<<<cgrd, cblk, 0, stream>>>((const float4*)wv, (uint64_t*)wvb, n4);
  f32_to_bf16_kernel<<<cgrd, cblk, 0, stream>>>((const float4*)wo, (uint64_t*)wob, n4);

  dim3 ggrd(SEQ / 128, HID / 128), gblk(256);
  gemm_bf16_wmma<<<ggrd, gblk, 0, stream>>>(xb, wqb, Qb, QSCALE, 0);
  gemm_bf16_wmma<<<ggrd, gblk, 0, stream>>>(xb, wkb, Kb, 1.0f, 0);
  gemm_bf16_wmma<<<ggrd, gblk, 0, stream>>>(xb, wvb, Vt, 1.0f, 1);

  flash_attn_kernel<<<dim3(SEQ / 64, NHEAD), 128, 0, stream>>>(Qb, Kb, Vt, Ab);

  gemm_bf16_wmma<<<ggrd, gblk, 0, stream>>>(Ab, wob, d_out, 1.0f, 2);
}
